// QuantizedLayer_79508434583604
// MI455X (gfx1250) — matrix operand order repaired and checked by reading, not compiled, not run on hardware
//
#include <hip/hip_runtime.h>
#include <hip/hip_bf16.h>
#include <stdint.h>

// Problem constants (match reference)
#define NB      8192      // batch
#define IN_DIM  4096
#define OUT_DIM 4096
#define NCHOICE 4
#define KSTEP   64

typedef __attribute__((ext_vector_type(16))) __bf16 bf16x16;
typedef __attribute__((ext_vector_type(2)))  __bf16 bf16x2;
typedef __attribute__((ext_vector_type(8)))  float  f32x8;

union FragCast { uint4 q[2]; bf16x16 v; };

// ---------------- helpers ----------------
// native conversion -> v_cvt_pk_bf16_f32 class hardware convert (RNE)
__device__ __forceinline__ unsigned pack2(float a, float b) {
    bf16x2 p;
    p.x = (__bf16)a;
    p.y = (__bf16)b;
    return __builtin_bit_cast(unsigned, p);
}
// monotone float<->uint mapping so atomicMin/Max on uint == float min/max
__device__ __forceinline__ unsigned encf(float f) {
    unsigned u = __float_as_uint(f);
    return (u & 0x80000000u) ? ~u : (u | 0x80000000u);
}
__device__ __forceinline__ float decf(unsigned e) {
    unsigned u = (e & 0x80000000u) ? (e & 0x7fffffffu) : ~e;
    return __uint_as_float(u);
}

// ---------------- workspace layout (uints) ----------------
// [0..3]   counts per choice (int)
// [4]      encoded global min of W
// [5]      encoded global max of W
// [16..]   bucket[NCHOICE][NB] sample indices (int)
#define WS_COUNTS 0
#define WS_MIN    4
#define WS_MAX    5
#define WS_BUCKET 16

__global__ void init_ws(unsigned* __restrict__ ws) {
    int t = threadIdx.x;
    if (t < NCHOICE) ((int*)ws)[WS_COUNTS + t] = 0;
    if (t == NCHOICE)     ws[WS_MIN] = 0xFFFFFFFFu;
    if (t == NCHOICE + 1) ws[WS_MAX] = 0u;
}

// ---------------- router: stats + MLP + gumbel argmax + bucket ----------------
__global__ __launch_bounds__(256)
void router_kernel(const float* __restrict__ x,
                   const float* __restrict__ gumbel,
                   const float* __restrict__ W1, const float* __restrict__ b1,
                   const float* __restrict__ W2, const float* __restrict__ b2,
                   float* __restrict__ out_logp, int* __restrict__ out_bits,
                   int* __restrict__ counts, int* __restrict__ bucket)
{
    __shared__ float s0[256], s1[256], s2[256];
    const int b = blockIdx.x;
    const int t = threadIdx.x;
    const float4* xr = (const float4*)(x + (size_t)b * IN_DIM);
    float sum = 0.f, sq = 0.f, z = 0.f;
#pragma unroll
    for (int i = 0; i < 4; ++i) {
        float4 v = xr[t + i * 256];
        sum += v.x + v.y + v.z + v.w;
        sq  += v.x * v.x + v.y * v.y + v.z * v.z + v.w * v.w;
        z   += (v.x == 0.f ? 1.f : 0.f) + (v.y == 0.f ? 1.f : 0.f)
             + (v.z == 0.f ? 1.f : 0.f) + (v.w == 0.f ? 1.f : 0.f);
    }
    s0[t] = sum; s1[t] = sq; s2[t] = z;
    __syncthreads();
    for (int off = 128; off > 0; off >>= 1) {
        if (t < off) { s0[t] += s0[t + off]; s1[t] += s1[t + off]; s2[t] += s2[t + off]; }
        __syncthreads();
    }
    if (t == 0) {
        const float n = (float)IN_DIM;
        float mean = s0[0] / n;
        float var  = (s1[0] - s0[0] * s0[0] / n) / (n - 1.0f);  // ddof=1
        float sp   = s2[0] / n;
        float st[3] = { mean, var, sp };
        float h[16];
#pragma unroll
        for (int j = 0; j < 16; ++j) {
            float a = b1[j];
#pragma unroll
            for (int k = 0; k < 3; ++k) a += st[k] * W1[j * 3 + k];
            h[j] = fmaxf(a, 0.f);
        }
        float logits[NCHOICE];
#pragma unroll
        for (int c = 0; c < NCHOICE; ++c) {
            float a = b2[c];
#pragma unroll
            for (int j = 0; j < 16; ++j) a += h[j] * W2[c * 16 + j];
            logits[c] = a;
        }
        const float* g = gumbel + (size_t)b * NCHOICE;
        int ci = 0; float best = logits[0] + g[0];
#pragma unroll
        for (int c = 1; c < NCHOICE; ++c) {
            float v = logits[c] + g[c];
            if (v > best) { best = v; ci = c; }   // first-max tie-break
        }
        float mx = logits[0];
#pragma unroll
        for (int c = 1; c < NCHOICE; ++c) mx = fmaxf(mx, logits[c]);
        float den = 0.f;
#pragma unroll
        for (int c = 0; c < NCHOICE; ++c) den += expf(logits[c] - mx);
        float p = expf(logits[ci] - mx) / den;
        out_logp[b] = logf(p + 1e-9f);
        const int bits_tab[NCHOICE] = { 4, 8, 16, 32 };
        out_bits[b] = bits_tab[ci];
        int pos = atomicAdd(&counts[ci], 1);
        bucket[ci * NB + pos] = b;
    }
}

// ---------------- W global min/max ----------------
__global__ __launch_bounds__(256)
void wminmax_kernel(const float* __restrict__ W, unsigned* __restrict__ ws)
{
    __shared__ unsigned smin[256], smax[256];
    const size_t N4 = (size_t)OUT_DIM * IN_DIM / 4;
    const float4* w4 = (const float4*)W;
    float lo = 3.4e38f, hi = -3.4e38f;
    for (size_t i = (size_t)blockIdx.x * 256 + threadIdx.x; i < N4;
         i += (size_t)gridDim.x * 256) {
        float4 v = w4[i];
        lo = fminf(lo, fminf(fminf(v.x, v.y), fminf(v.z, v.w)));
        hi = fmaxf(hi, fmaxf(fmaxf(v.x, v.y), fmaxf(v.z, v.w)));
    }
    const int t = threadIdx.x;
    smin[t] = encf(lo); smax[t] = encf(hi);
    __syncthreads();
    for (int off = 128; off > 0; off >>= 1) {
        if (t < off) {
            smin[t] = min(smin[t], smin[t + off]);
            smax[t] = max(smax[t], smax[t + off]);
        }
        __syncthreads();
    }
    if (t == 0) {
        atomicMin(&ws[WS_MIN], smin[0]);
        atomicMax(&ws[WS_MAX], smax[0]);
    }
}

// ---------------- grouped GEMM: out = relu(x[rows] @ fq(W,bits).T + b) ----------------
// grid: (OUT_DIM/128, NB/64, NCHOICE), block: 256 threads = 8 waves
// wave grid 2(M) x 4(N); each wave -> 32x32 via 2x2 v_wmma_f32_16x16x32_bf16
// K staged in 64-wide LDS tiles -> 8 wmma per loop body, 2 barriers per 64 K
__global__ __launch_bounds__(256)
void gemm_kernel(const float* __restrict__ x, const float* __restrict__ W,
                 const float* __restrict__ bias, float* __restrict__ out,
                 const int* __restrict__ counts, const int* __restrict__ bucket,
                 const unsigned* __restrict__ ws)
{
    const int c   = blockIdx.z;
    const int cnt = counts[c];
    const int m0  = blockIdx.y * 64;
    if (m0 >= cnt) return;
    const int n0  = blockIdx.x * 128;

    __shared__ int            s_rows[64];
    __shared__ unsigned short sA[64 * KSTEP];    // [row][k] bf16, stride 64
    __shared__ unsigned short sB[128 * KSTEP];   // [n][k]   bf16, stride 64

    const int t = threadIdx.x;

    // fake-quant parameters (uniform per block)
    const float mn = decf(ws[WS_MIN]);
    const float mx = decf(ws[WS_MAX]);
    const int bits_tab[NCHOICE] = { 4, 8, 16, 32 };
    const int bits = bits_tab[c];
    const bool ident = (bits == 32);
    float scale = 0.f, rs = 0.f;
    if (!ident) {
        float ql = (float)((1u << bits) - 1u);       // 2^bits - 1
        scale = (mx - mn) / (ql + 1e-9f);
        rs = 1.0f / (scale + 1e-9f);
    }

    if (t < 64) {
        int m = m0 + t;
        s_rows[t] = (m < cnt) ? bucket[c * NB + m] : bucket[c * NB];  // safe fallback
    }
    __syncthreads();

    // loader roles
    const int ar = t >> 2, ak = (t & 3) * 16;    // A: 4 thr/row, 16 k each
    const int bn = t >> 1, bk = (t & 1) * 32;    // B: 2 thr/row, 32 k each
    const size_t xbase = (size_t)s_rows[ar] * IN_DIM + ak;
    const size_t wbase = (size_t)(n0 + bn) * IN_DIM + bk;

    // wave decomposition
    const int wave = t >> 5, lane = t & 31;
    const int wm = wave >> 2, wn = wave & 3;
    const int lr = lane & 15, lhi = lane >> 4;
    const int klo = lhi ? 8 : 0;    // A: lanes<16 -> K 0-7 & 16-23; lanes>=16 -> K 8-15 & 24-31
    const int kb2 = lhi ? 8 : 0;    // B: lanes<16 -> K 0-7 & 16-23; lanes>=16 -> K 8-15 & 24-31

    const f32x8 zero8 = { 0.f, 0.f, 0.f, 0.f, 0.f, 0.f, 0.f, 0.f };
    f32x8 acc[2][2] = { { zero8, zero8 }, { zero8, zero8 } };

    for (int k0 = 0; k0 < IN_DIM; k0 += KSTEP) {
        // ---- stage A tile (row-gathered x, fp32 -> bf16), 16 k per thread ----
        {
            const float4* xp = (const float4*)(x + xbase + k0);
            float4 a0 = xp[0], a1 = xp[1], a2 = xp[2], a3 = xp[3];
            uint4 p0 = { pack2(a0.x, a0.y), pack2(a0.z, a0.w),
                         pack2(a1.x, a1.y), pack2(a1.z, a1.w) };
            uint4 p1 = { pack2(a2.x, a2.y), pack2(a2.z, a2.w),
                         pack2(a3.x, a3.y), pack2(a3.z, a3.w) };
            *(uint4*)&sA[ar * KSTEP + ak]     = p0;
            *(uint4*)&sA[ar * KSTEP + ak + 8] = p1;
        }

        // ---- stage B tile (W, fake-quantized on the fly, fp32 -> bf16), 32 k ----
#pragma unroll
        for (int h = 0; h < 2; ++h) {
            const float4* wp = (const float4*)(W + wbase + k0 + h * 16);
            float wv[16];
#pragma unroll
            for (int q = 0; q < 4; ++q) {
                float4 w4v = wp[q];
                wv[q * 4 + 0] = w4v.x; wv[q * 4 + 1] = w4v.y;
                wv[q * 4 + 2] = w4v.z; wv[q * 4 + 3] = w4v.w;
            }
            if (!ident) {
#pragma unroll
                for (int q = 0; q < 16; ++q)
                    wv[q] = rintf((wv[q] - mn) * rs) * scale + mn;
            }
            uint4 p0 = { pack2(wv[0],  wv[1]),  pack2(wv[2],  wv[3]),
                         pack2(wv[4],  wv[5]),  pack2(wv[6],  wv[7]) };
            uint4 p1 = { pack2(wv[8],  wv[9]),  pack2(wv[10], wv[11]),
                         pack2(wv[12], wv[13]), pack2(wv[14], wv[15]) };
            *(uint4*)&sB[bn * KSTEP + bk + h * 16]     = p0;
            *(uint4*)&sB[bn * KSTEP + bk + h * 16 + 8] = p1;
        }

        __syncthreads();

        // ---- two 32-deep WMMA passes over the staged 64-wide tile ----
#pragma unroll
        for (int kc = 0; kc < 2; ++kc) {
            bf16x16 afrag[2], bfrag[2];
#pragma unroll
            for (int mi = 0; mi < 2; ++mi) {
                int base = (wm * 32 + mi * 16 + lr) * KSTEP + kc * 32 + klo;
                FragCast fc;
                fc.q[0] = *(const uint4*)&sA[base];
                fc.q[1] = *(const uint4*)&sA[base + 16];
                afrag[mi] = fc.v;
            }
#pragma unroll
            for (int ni = 0; ni < 2; ++ni) {
                int base = (wn * 32 + ni * 16 + lr) * KSTEP + kc * 32 + kb2;
                FragCast fc;
                fc.q[0] = *(const uint4*)&sB[base];
                fc.q[1] = *(const uint4*)&sB[base + 16];
                bfrag[ni] = fc.v;
            }
#pragma unroll
            for (int mi = 0; mi < 2; ++mi)
#pragma unroll
                for (int ni = 0; ni < 2; ++ni)
                    acc[mi][ni] = __builtin_amdgcn_wmma_f32_16x16x32_bf16(
                        false, afrag[mi], false, bfrag[ni],
                        (short)0, acc[mi][ni], false, false);
        }

        __syncthreads();
    }

    // ---- epilogue: +bias, relu, scatter by bucketed row ----
#pragma unroll
    for (int ni = 0; ni < 2; ++ni) {
        const int col = n0 + wn * 32 + ni * 16 + lr;
        const float bv = bias[col];
#pragma unroll
        for (int mi = 0; mi < 2; ++mi) {
            f32x8 av = acc[mi][ni];
#pragma unroll
            for (int v = 0; v < 8; ++v) {
                int rt = wm * 32 + mi * 16 + v + lhi * 8;   // C/D: VGPR v -> M=v(+8)
                int m = m0 + rt;
                if (m < cnt) {
                    float r = fmaxf(av[v] + bv, 0.f);
                    out[(size_t)s_rows[rt] * OUT_DIM + col] = r;
                }
            }
        }
    }
}

// ---------------- launcher ----------------
extern "C" void kernel_launch(void* const* d_in, const int* in_sizes, int n_in,
                              void* d_out, int out_size, void* d_ws, size_t ws_size,
                              hipStream_t stream)
{
    const float* x      = (const float*)d_in[0];
    // d_in[1] = temp (unused: tau only rescales, argmax unaffected)
    const float* gumbel = (const float*)d_in[2];
    const float* W_lin  = (const float*)d_in[3];
    const float* b_lin  = (const float*)d_in[4];
    const float* W1     = (const float*)d_in[5];
    const float* b1     = (const float*)d_in[6];
    const float* W2     = (const float*)d_in[7];
    const float* b2     = (const float*)d_in[8];

    float* out_main = (float*)d_out;                        // [NB, OUT_DIM]
    float* out_logp = out_main + (size_t)NB * OUT_DIM;      // [NB]
    int*   out_bits = (int*)(out_logp + NB);                // [NB] (int32)

    unsigned* ws     = (unsigned*)d_ws;
    int*      counts = (int*)ws + WS_COUNTS;
    int*      bucket = (int*)ws + WS_BUCKET;

    init_ws<<<1, 64, 0, stream>>>(ws);

    router_kernel<<<NB, 256, 0, stream>>>(x, gumbel, W1, b1, W2, b2,
                                          out_logp, out_bits, counts, bucket);

    wminmax_kernel<<<512, 256, 0, stream>>>(W_lin, ws);

    dim3 grid(OUT_DIM / 128, NB / 64, NCHOICE);
    gemm_kernel<<<grid, 256, 0, stream>>>(x, W_lin, b_lin, out_main,
                                          counts, bucket, ws);
}
